// RbfNet_17523466568133
// MI455X (gfx1250) — matrix operand order repaired and checked by reading, not compiled, not run on hardware
//
#include <hip/hip_runtime.h>

typedef __attribute__((ext_vector_type(16))) __bf16 v16bf;
typedef __attribute__((ext_vector_type(8)))  float  v8f;

struct EdgeRec { int src; int tgt; int cell; int pad0; float w[4]; }; // 32 B

// ---------- helpers ----------
__device__ __forceinline__ unsigned short f2bf(float f) {
  unsigned u = __float_as_uint(f);
  u += 0x7FFFu + ((u >> 16) & 1u);          // round-to-nearest-even
  return (unsigned short)(u >> 16);
}

__device__ __forceinline__ void gatomic_add_f32(float* p, float v) {
  // single hardware atomic (no CAS loop): GLOBAL_ATOMIC_ADD_F32, no return
  asm volatile("global_atomic_add_f32 %0, %1, off" : : "v"(p), "v"(v) : "memory");
}

// tent-basis cell + 4 corner weights for one edge direction (already clipped)
__device__ __forceinline__ int edge_cell(float dx, float dy, float w[4]) {
  const float SP = 2.0f / 7.0f, INV = 3.5f;
  float r = sqrtf(dx * dx + dy * dy + 1e-12f);
  float u = 2.0f * r - 1.0f;
  float v = atan2f(dy, dx) * 0.3183098861837907f; // theta / pi
  int iu = (int)floorf((u + 1.0f) * INV); iu = max(0, min(6, iu));
  int iv = (int)floorf((v + 1.0f) * INV); iv = max(0, min(6, iv));
  float cu = -1.0f + iu * SP, cv = -1.0f + iv * SP;
  float wu0 = fmaxf(0.f, 1.f - fabsf(u - cu) * INV);
  float wu1 = fmaxf(0.f, 1.f - fabsf(u - cu - SP) * INV);
  float wv0 = fmaxf(0.f, 1.f - fabsf(v - cv) * INV);
  float wv1 = fmaxf(0.f, 1.f - fabsf(v - cv - SP) * INV);
  w[0] = wu0 * wv0; w[1] = wu0 * wv1; w[2] = wu1 * wv0; w[3] = wu1 * wv1;
  return iu * 7 + iv;
}

__device__ __forceinline__ void edge_dir(const float* posT, const float* posS,
                                         int tgt, int src, float sup, float sgn,
                                         float& dx, float& dy) {
  dx = sgn * (posS[2 * src]     - posT[2 * tgt])     / sup;
  dy = sgn * (posS[2 * src + 1] - posT[2 * tgt + 1]) / sup;
  dx = fminf(1.f, fmaxf(-1.f, dx));
  dy = fminf(1.f, fmaxf(-1.f, dy));
}

// ---------- small prep kernels ----------
__global__ void k_zero_i32(int* p, int n) {
  int i = blockIdx.x * blockDim.x + threadIdx.x;
  if (i < n) p[i] = 0;
}

__global__ void k_pack_feats(const float* __restrict__ src, unsigned short* __restrict__ dst,
                             int n, int cin) { // dst stride 32, zero padded
  int idx = blockIdx.x * blockDim.x + threadIdx.x;
  if (idx >= n * 32) return;
  int node = idx >> 5, c = idx & 31;
  dst[idx] = f2bf(c < cin ? src[node * cin + c] : 0.0f);
}

// repack W[k][cin][cout] f32 -> bf16 fragment-native tiles:
// Wf[((k*chunks+ch)*blks+b)*512 + lane*16 + e]; lane<16 -> K=ch*32+e, lane>=16 -> K=ch*32+16+e; N=b*16+(lane&15)
__global__ void k_pack_w(const float* __restrict__ W, unsigned short* __restrict__ Wf,
                         int cinOrig, int coutOrig, int chunks, int blks) {
  int idx = blockIdx.x * blockDim.x + threadIdx.x;
  int total = 64 * chunks * blks * 512;
  if (idx >= total) return;
  int e = idx & 15, lane = (idx >> 4) & 31, r = idx >> 9;
  int b = r % blks; r /= blks;
  int ch = r % chunks;
  int k = r / chunks;
  int K = ch * 32 + ((lane < 16) ? 0 : 16) + e;
  int N = b * 16 + (lane & 15);
  float v = (K < cinOrig && N < coutOrig) ? W[(size_t)(k * cinOrig + K) * coutOrig + N] : 0.0f;
  Wf[idx] = f2bf(v);
}

// ---------- binning ----------
__global__ void k_hist(const float* __restrict__ posT, const float* __restrict__ posS,
                       const int* __restrict__ ti, const int* __restrict__ si,
                       const float* __restrict__ sup, float sgn, int E, int* __restrict__ hist) {
  int e = blockIdx.x * blockDim.x + threadIdx.x;
  if (e >= E) return;
  float dx, dy, w[4];
  edge_dir(posT, posS, ti[e], si[e], sup[0], sgn, dx, dy);
  atomicAdd(&hist[edge_cell(dx, dy, w)], 1);
}

__global__ void k_scan(const int* histF, int* startF, int* cursF,
                       const int* histB, int* startB, int* cursB) {
  if (threadIdx.x == 0) {
    int s = 0;
    for (int c = 0; c < 49; ++c) { startF[c] = s; cursF[c] = s; s += (histF[c] + 15) & ~15; }
    startF[49] = s;
  } else if (threadIdx.x == 1) {
    int s = 0;
    for (int c = 0; c < 49; ++c) { startB[c] = s; cursB[c] = s; s += (histB[c] + 15) & ~15; }
    startB[49] = s;
  }
}

__global__ void k_scatter(const float* __restrict__ posT, const float* __restrict__ posS,
                          const int* __restrict__ ti, const int* __restrict__ si,
                          const float* __restrict__ sup, float sgn, int E,
                          int* __restrict__ curs, EdgeRec* __restrict__ recs) {
  int e = blockIdx.x * blockDim.x + threadIdx.x;
  if (e >= E) return;
  float dx, dy, w[4];
  edge_dir(posT, posS, ti[e], si[e], sup[0], sgn, dx, dy);
  int cell = edge_cell(dx, dy, w);
  int pos = atomicAdd(&curs[cell], 1);
  EdgeRec r; r.src = si[e]; r.tgt = ti[e]; r.cell = cell; r.pad0 = 0;
  r.w[0] = w[0]; r.w[1] = w[1]; r.w[2] = w[2]; r.w[3] = w[3];
  recs[pos] = r;
}

// ---------- dense shortcut layers ----------
__global__ void k_lin(const float* __restrict__ ff, const float* __restrict__ fc0,
                      float* __restrict__ ans1, int NF) { // init ans1[NF,96]
  int idx = blockIdx.x * blockDim.x + threadIdx.x;
  if (idx >= NF * 96) return;
  int node = idx / 96, col = idx % 96;
  float v = 0.f;
  if (col < 32) {
    #pragma unroll
    for (int i = 0; i < 4; ++i) v += ff[node * 4 + i] * fc0[i * 32 + col];
    v = fmaxf(v, 0.f);
  }
  ans1[idx] = v;
}

__global__ void k_finalize(float* __restrict__ a, unsigned short* __restrict__ ab, int n) {
  int i = blockIdx.x * blockDim.x + threadIdx.x;
  if (i >= n) return;
  float v = fmaxf(a[i], 0.f);
  a[i] = v; ab[i] = f2bf(v);
}

__global__ void k_fc(const float* __restrict__ x, const float* __restrict__ w,
                     float* __restrict__ y, int n, int cin, int cout, int addRes) {
  int idx = blockIdx.x * blockDim.x + threadIdx.x;
  if (idx >= n * cout) return;
  int node = idx / cout, co = idx % cout;
  float acc = addRes ? x[node * cin + co] : 0.f; // residual (cin==cout)
  for (int i = 0; i < cin; ++i) acc += x[node * cin + i] * w[i * cout + co];
  y[idx] = acc;
}

// ---------- WMMA edge convolution ----------
// one wave per 16-edge tile (all edges share a basis cell -> same 4 W_k).
template <int CHUNKS, int BLKS>
__global__ void k_conv(const EdgeRec* __restrict__ recs, int nTiles,
                       const unsigned short* __restrict__ X,   // bf16 [*, CHUNKS*32]
                       const unsigned short* __restrict__ Wf,  // bf16 fragment tiles
                       float* __restrict__ out, int outStride, int outOffset, int outCols) {
  int tile = blockIdx.x * (blockDim.x >> 5) + (threadIdx.x >> 5);
  if (tile >= nTiles) return;              // wave-uniform: EXEC stays all-ones for WMMA
  const int lane = threadIdx.x & 31;
  const int hi = (lane >= 16);
  const EdgeRec* rec = recs + (size_t)tile * 16;

  int cell = rec[0].cell;
  int iu = cell / 7, iv = cell % 7;
  int kc[4] = { iu * 8 + iv, iu * 8 + iv + 1, (iu + 1) * 8 + iv, (iu + 1) * 8 + iv + 1 };

  const unsigned short* Xrow = X + (size_t)rec[lane & 15].src * (CHUNKS * 32);

  v8f acc[4][BLKS] = {};
  #pragma unroll
  for (int ch = 0; ch < CHUNKS; ++ch) {
    union { v16bf v; uint4 q[2]; } A;      // 16-bit A 16x32: lane<16 K=0..15, lane>=16 K=16..31
    int offA = ch * 32 + (hi ? 16 : 0);
    A.q[0] = *(const uint4*)(Xrow + offA);
    A.q[1] = *(const uint4*)(Xrow + offA + 8);
    #pragma unroll
    for (int c = 0; c < 4; ++c) {
      const unsigned short* Wk = Wf + (size_t)((kc[c] * CHUNKS + ch) * BLKS) * 512;
      #pragma unroll
      for (int b = 0; b < BLKS; ++b) {
        union { v16bf v; uint4 q[2]; } B;
        const unsigned short* wb = Wk + b * 512 + lane * 16;
        B.q[0] = *(const uint4*)(wb);
        B.q[1] = *(const uint4*)(wb + 8);
        acc[c][b] = __builtin_amdgcn_wmma_f32_16x16x32_bf16(
            false, A.v, false, B.v, (short)0, acc[c][b], false, false);
      }
    }
  }

  // epilogue: combine 4 corner accumulators with per-row weights, scatter-add
  int col = lane & 15;
  #pragma unroll
  for (int r = 0; r < 8; ++r) {
    const EdgeRec& er = rec[r + (hi ? 8 : 0)];  // C/D row M = r (+8 for upper half)
    float w0 = er.w[0], w1 = er.w[1], w2 = er.w[2], w3 = er.w[3];
    float* orow = out + (size_t)er.tgt * outStride + outOffset;
    #pragma unroll
    for (int b = 0; b < BLKS; ++b) {
      int cc = b * 16 + col;
      if (cc < outCols) {
        float val = w0 * acc[0][b][r] + w1 * acc[1][b][r] +
                    w2 * acc[2][b][r] + w3 * acc[3][b][r];
        gatomic_add_f32(orow + cc, val);
      }
    }
  }
}

// ---------- host ----------
static inline int cdiv(int a, int b) { return (a + b - 1) / b; }

extern "C" void kernel_launch(void* const* d_in, const int* in_sizes, int n_in,
                              void* d_out, int out_size, void* d_ws, size_t ws_size,
                              hipStream_t stream) {
  const float* fp    = (const float*)d_in[0];
  const float* bp    = (const float*)d_in[1];
  const float* ff    = (const float*)d_in[2];
  const float* bfeat = (const float*)d_in[3];
  const float* sup   = (const float*)d_in[4];
  const int*   fi    = (const int*)d_in[5];
  const int*   fj    = (const int*)d_in[6];
  const int*   bfi   = (const int*)d_in[7];
  const int*   bbi   = (const int*)d_in[8];
  const float* W0 = (const float*)d_in[9];
  const float* W1 = (const float*)d_in[10];
  const float* W2 = (const float*)d_in[11];
  const float* W3 = (const float*)d_in[12];
  const float* W4 = (const float*)d_in[13];
  const float* fc0 = (const float*)d_in[14];
  const float* fc1 = (const float*)d_in[15];
  const float* fc2 = (const float*)d_in[16];
  const float* fc3 = (const float*)d_in[17];

  const int NF = in_sizes[0] / 2, NB = in_sizes[1] / 2;
  const int EF = in_sizes[5],     EB = in_sizes[7];
  float* out = (float*)d_out;

  char* ws = (char*)d_ws; size_t off = 0;
  auto alloc = [&](size_t bytes) -> void* {
    void* p = ws + off; off = (off + bytes + 255) & ~(size_t)255; return p;
  };

  unsigned short* Xf0 = (unsigned short*)alloc((size_t)NF * 32 * 2);
  unsigned short* Xb0 = (unsigned short*)alloc((size_t)NB * 32 * 2);
  unsigned short* Wb0 = (unsigned short*)alloc(64 * 1 * 2 * 512 * 2);
  unsigned short* Wb1 = (unsigned short*)alloc(64 * 1 * 2 * 512 * 2);
  unsigned short* Wb2 = (unsigned short*)alloc(64 * 3 * 4 * 512 * 2);
  unsigned short* Wb3 = (unsigned short*)alloc(64 * 2 * 4 * 512 * 2);
  unsigned short* Wb4 = (unsigned short*)alloc(64 * 2 * 1 * 512 * 2);
  float*          ans1  = (float*)alloc((size_t)NF * 96 * 4);
  unsigned short* ans1b = (unsigned short*)alloc((size_t)NF * 96 * 2);
  float*          ans2  = (float*)alloc((size_t)NF * 64 * 4);
  unsigned short* ans2b = (unsigned short*)alloc((size_t)NF * 64 * 2);
  float*          ans3  = (float*)alloc((size_t)NF * 64 * 4);
  unsigned short* ans3b = (unsigned short*)alloc((size_t)NF * 64 * 2);
  const int tilesF = cdiv(EF, 16) + 49;
  const int tilesB = cdiv(EB, 16) + 49;
  EdgeRec* recF = (EdgeRec*)alloc((size_t)tilesF * 16 * sizeof(EdgeRec));
  EdgeRec* recB = (EdgeRec*)alloc((size_t)tilesB * 16 * sizeof(EdgeRec));
  int* ints = (int*)alloc(512 * 4);
  int* histF = ints;       int* startF = ints + 64;  int* cursF = ints + 128;
  int* histB = ints + 192; int* startB = ints + 256; int* cursB = ints + 320;

  const int T = 256;
  // zero hist/start/curs + zero edge records (pad records must be zero-weight)
  k_zero_i32<<<cdiv(512, T), T, 0, stream>>>(ints, 512);
  k_zero_i32<<<cdiv(tilesF * 16 * 8, T), T, 0, stream>>>((int*)recF, tilesF * 16 * 8);
  k_zero_i32<<<cdiv(tilesB * 16 * 8, T), T, 0, stream>>>((int*)recB, tilesB * 16 * 8);

  // pack features & weights to bf16
  k_pack_feats<<<cdiv(NF * 32, T), T, 0, stream>>>(ff, Xf0, NF, 4);
  k_pack_feats<<<cdiv(NB * 32, T), T, 0, stream>>>(bfeat, Xb0, NB, 4);
  k_pack_w<<<cdiv(64 * 1 * 2 * 512, T), T, 0, stream>>>(W0, Wb0, 4, 32, 1, 2);
  k_pack_w<<<cdiv(64 * 1 * 2 * 512, T), T, 0, stream>>>(W1, Wb1, 4, 32, 1, 2);
  k_pack_w<<<cdiv(64 * 3 * 4 * 512, T), T, 0, stream>>>(W2, Wb2, 96, 64, 3, 4);
  k_pack_w<<<cdiv(64 * 2 * 4 * 512, T), T, 0, stream>>>(W3, Wb3, 64, 64, 2, 4);
  k_pack_w<<<cdiv(64 * 2 * 1 * 512, T), T, 0, stream>>>(W4, Wb4, 64, 2, 2, 1);

  // bin edges by basis cell (fluid: d = -(fp[fj]-fp[fi]) -> sgn=-1; boundary: sgn=+1)
  k_hist<<<cdiv(EF, T), T, 0, stream>>>(fp, fp, fi, fj, sup, -1.f, EF, histF);
  k_hist<<<cdiv(EB, T), T, 0, stream>>>(fp, bp, bfi, bbi, sup, 1.f, EB, histB);
  k_scan<<<1, 32, 0, stream>>>(histF, startF, cursF, histB, startB, cursB);
  k_scatter<<<cdiv(EF, T), T, 0, stream>>>(fp, fp, fi, fj, sup, -1.f, EF, cursF, recF);
  k_scatter<<<cdiv(EB, T), T, 0, stream>>>(fp, bp, bfi, bbi, sup, 1.f, EB, cursB, recB);

  // layer 1: lin | fconv | bconv  -> ans1[NF,96]
  k_lin<<<cdiv(NF * 96, T), T, 0, stream>>>(ff, fc0, ans1, NF);
  k_conv<1, 2><<<cdiv(tilesF, 8), 256, 0, stream>>>(recF, tilesF, Xf0, Wb0, ans1, 96, 32, 32);
  k_conv<1, 2><<<cdiv(tilesB, 8), 256, 0, stream>>>(recB, tilesB, Xb0, Wb1, ans1, 96, 64, 32);
  k_finalize<<<cdiv(NF * 96, T), T, 0, stream>>>(ans1, ans1b, NF * 96);

  // layer 2: relu(conv(ans1) + ans1@fc1)
  k_fc<<<cdiv(NF * 64, T), T, 0, stream>>>(ans1, fc1, ans2, NF, 96, 64, 0);
  k_conv<3, 4><<<cdiv(tilesF, 8), 256, 0, stream>>>(recF, tilesF, ans1b, Wb2, ans2, 64, 0, 64);
  k_finalize<<<cdiv(NF * 64, T), T, 0, stream>>>(ans2, ans2b, NF * 64);

  // layer 3: relu(conv(ans2) + ans2@fc2 + ans2)
  k_fc<<<cdiv(NF * 64, T), T, 0, stream>>>(ans2, fc2, ans3, NF, 64, 64, 1);
  k_conv<2, 4><<<cdiv(tilesF, 8), 256, 0, stream>>>(recF, tilesF, ans2b, Wb3, ans3, 64, 0, 64);
  k_finalize<<<cdiv(NF * 64, T), T, 0, stream>>>(ans3, ans3b, NF * 64);

  // layer 4 (no relu, no residual): conv(ans3) + ans3@fc3 -> out[NF,2]
  k_fc<<<cdiv(NF * 2, T), T, 0, stream>>>(ans3, fc3, out, NF, 64, 2, 0);
  k_conv<2, 1><<<cdiv(tilesF, 8), 256, 0, stream>>>(recF, tilesF, ans3b, Wb4, out, 2, 0, 2);
}
